// HTDM_32014686224787
// MI455X (gfx1250) — compile-verified, matrix operand order checked by reading, not run on hardware
//
#include <hip/hip_runtime.h>

typedef __attribute__((ext_vector_type(16))) __bf16 v16bf;
typedef __attribute__((ext_vector_type(8)))  __bf16 v8bf;
typedef __attribute__((ext_vector_type(4)))  __bf16 v4bf;
typedef __attribute__((ext_vector_type(8)))  float  v8f;
typedef __attribute__((ext_vector_type(4)))  float  v4f;
typedef __attribute__((ext_vector_type(4)))  int    v4i;

#define DEVI __device__ __forceinline__

#if defined(__has_builtin)
#  if __has_builtin(__builtin_amdgcn_global_load_async_to_lds_b128)
#    define HAVE_ASYNC_LDS 1
#  endif
#endif
#ifndef HAVE_ASYNC_LDS
#  define HAVE_ASYNC_LDS 0
#endif

constexpr int Bb = 8, Nn = 243, Jj = 17, Cc = 512, Hh = 8, Dd = 64;
constexpr int RROWS = Bb * Jj * Nn;          // 33048 rows in (b,j,n) order
constexpr int MT    = (RROWS + 15) / 16;     // 2066 M-tiles
constexpr int NP    = 256;                   // padded N for attention K-dim

// children of each joint (leaf -> itself), padded to 3
__device__ __constant__ int   d_child[17][3] = {
  {1,4,7},{2,2,2},{3,3,3},{3,3,3},{5,5,5},{6,6,6},{6,6,6},{8,8,8},
  {9,11,14},{10,10,10},{10,10,10},{12,12,12},{13,13,13},{13,13,13},
  {15,15,15},{16,16,16},{16,16,16}};
__device__ __constant__ int   d_ccnt[17] = {3,1,1,1,1,1,1,1,3,1,1,1,1,1,1,1,1};
__device__ __constant__ float d_cinv[17] = {1.f/3.f,1,1,1,1,1,1,1,1.f/3.f,1,1,1,1,1,1,1,1};

#if HAVE_ASYNC_LDS
// 16-byte async global->LDS copy (tracked by ASYNCcnt)
DEVI void async_cp16(const void* g, void* l) {
  void* gp = const_cast<void*>(g);
  __builtin_amdgcn_global_load_async_to_lds_b128(
      (__attribute__((address_space(1))) v4i*)gp,
      (__attribute__((address_space(3))) v4i*)l, 0, 0);
}
#endif

// Fragment = two contiguous 16B chunks: elements i=0..7 at p[0..7], i=8..15 at p[16..23]
DEVI v16bf frag16(const __bf16* p) {
  v8bf lo = *(const v8bf*)p;
  v8bf hi = *(const v8bf*)(p + 16);
  return __builtin_shufflevector(lo, hi, 0,1,2,3,4,5,6,7,8,9,10,11,12,13,14,15);
}

DEVI v8f wmma_bf16(v16bf a, v16bf b, v8f c) {
  return __builtin_amdgcn_wmma_f32_16x16x32_bf16(false, a, false, b, (short)0, c, false, false);
}

DEVI void row_decomp(int gr, int& b, int& j, int& n) {
  b = gr / (Jj * Nn);
  int r = gr % (Jj * Nn);
  j = r / Nn;
  n = r % Nn;
}

// ---------------------------------------------------------------------------
// Weight transpose+convert: Wt[c*512 + k] = bf16(W[k*ncols + c])
// ---------------------------------------------------------------------------
__global__ __launch_bounds__(256) void k_wt(const float* __restrict__ W,
                                            __bf16* __restrict__ Wt, int ncols) {
  int idx = blockIdx.x * 256 + threadIdx.x;
  if (idx >= ncols * 512) return;
  int k = idx / ncols, c = idx % ncols;
  Wt[(size_t)c * 512 + k] = (__bf16)W[idx];
}

// ---------------------------------------------------------------------------
// Kernel 1: row(r) @ [W_qkv | W_q] + bias -> qt/kt (B,J,H,N,D), v transposed
// into vtr (B,J,H,D,NP), qh (B,J,H,N,D).  Wt holds cols 0..2047.
// ---------------------------------------------------------------------------
__global__ __launch_bounds__(256) void k_qkv(
    const float* __restrict__ x, const __bf16* __restrict__ Wt,
    const float* __restrict__ bqkv, const float* __restrict__ bq,
    __bf16* __restrict__ qt, __bf16* __restrict__ kt,
    __bf16* __restrict__ vtr, __bf16* __restrict__ qh)
{
  __shared__ __bf16 ldsA[16 * 512];
  __shared__ int sXo[16], sB[16], sJ[16], sN[16];
  const int tid = threadIdx.x, lane = tid & 31, w = tid >> 5;
  const int nl = lane & 15, g = lane >> 4;
  const int mtile = blockIdx.x;

  if (tid < 16) {
    int gr = min(16 * mtile + tid, RROWS - 1);
    int b, j, n; row_decomp(gr, b, j, n);
    sXo[tid] = (b * Nn + n) * Jj;  // x row group (add joint, then *Cc)
    sB[tid] = b; sJ[tid] = j; sN[tid] = n;
  }
  __syncthreads();

  for (int idx = tid; idx < 16 * 128; idx += 256) {
    int r = idx >> 7, k4 = (idx & 127) << 2;
    const float4 v = *(const float4*)&x[(size_t)(sXo[r] + sJ[r]) * Cc + k4];
    v4bf o = { (__bf16)v.x, (__bf16)v.y, (__bf16)v.z, (__bf16)v.w };
    *(v4bf*)&ldsA[r * 512 + k4] = o;
  }
  __syncthreads();

  const int col0 = (blockIdx.y * 8 + w) * 16;
  const __bf16* wcol = Wt + (size_t)(col0 + nl) * 512;
  const __bf16* arow = ldsA + nl * 512;

  v8f acc = {};
#pragma unroll 4
  for (int ks = 0; ks < 16; ++ks) {
    const int off = ks * 32 + 8 * g;
    acc = wmma_bf16(frag16(arow + off), frag16(wcol + off), acc);
  }

  const bool isq = (col0 >= 1536);
  const int  c   = col0 + nl;
  const float bv = isq ? bq[c - 1536] : bqkv[c];
#pragma unroll
  for (int i = 0; i < 8; ++i) {
    int m = i + 8 * g;
    if (16 * mtile + m >= RROWS) continue;
    const int bjh = (sB[m] * Jj + sJ[m]) * Hh;  // head added below
    const int n   = sN[m];
    float v = acc[i] + bv;
    if (!isq) {
      int t = c >> 9, h = (c >> 6) & 7, d = c & 63;
      if (t == 2) {
        vtr[((size_t)(bjh + h) * Dd + d) * NP + n] = (__bf16)v;
      } else {
        __bf16* dst = (t == 0) ? qt : kt;
        dst[((size_t)(bjh + h) * Nn + n) * Dd + d] = (__bf16)v;
      }
    } else {
      int ch = c - 1536, h = ch >> 6, d = ch & 63;
      qh[((size_t)(bjh + h) * Nn + n) * Dd + d] = (__bf16)v;
    }
  }
}

// ---------------------------------------------------------------------------
// Kernel 2: child-mean rows @ W_kv + b_kv -> kh (B,J,H,N,D), vhr (B,J,H,D,NP)
// ---------------------------------------------------------------------------
__global__ __launch_bounds__(256) void k_kv(
    const float* __restrict__ x, const __bf16* __restrict__ Wt,
    const float* __restrict__ bkv,
    __bf16* __restrict__ kh, __bf16* __restrict__ vhr)
{
  __shared__ __bf16 ldsA[16 * 512];
  __shared__ int sXo[16], sB[16], sJ[16], sN[16];
  const int tid = threadIdx.x, lane = tid & 31, w = tid >> 5;
  const int nl = lane & 15, g = lane >> 4;
  const int mtile = blockIdx.x;

  if (tid < 16) {
    int gr = min(16 * mtile + tid, RROWS - 1);
    int b, j, n; row_decomp(gr, b, j, n);
    sXo[tid] = (b * Nn + n) * Jj;
    sB[tid] = b; sJ[tid] = j; sN[tid] = n;
  }
  __syncthreads();

  for (int idx = tid; idx < 16 * 128; idx += 256) {
    int r = idx >> 7, k4 = (idx & 127) << 2;
    const int j = sJ[r];
    const size_t rb = (size_t)sXo[r] * Cc + k4;
    int cnt = d_ccnt[j];
    float4 s = *(const float4*)&x[rb + (size_t)d_child[j][0] * Cc];
    for (int ci = 1; ci < cnt; ++ci) {
      const float4 t = *(const float4*)&x[rb + (size_t)d_child[j][ci] * Cc];
      s.x += t.x; s.y += t.y; s.z += t.z; s.w += t.w;
    }
    const float iv = d_cinv[j];
    v4bf o = { (__bf16)(s.x*iv), (__bf16)(s.y*iv), (__bf16)(s.z*iv), (__bf16)(s.w*iv) };
    *(v4bf*)&ldsA[r * 512 + k4] = o;
  }
  __syncthreads();

  const int col0 = (blockIdx.y * 8 + w) * 16;
  const __bf16* wcol = Wt + (size_t)(col0 + nl) * 512;
  const __bf16* arow = ldsA + nl * 512;

  v8f acc = {};
#pragma unroll 4
  for (int ks = 0; ks < 16; ++ks) {
    const int off = ks * 32 + 8 * g;
    acc = wmma_bf16(frag16(arow + off), frag16(wcol + off), acc);
  }

  const int   c  = col0 + nl;
  const float bv = bkv[c];
#pragma unroll
  for (int i = 0; i < 8; ++i) {
    int m = i + 8 * g;
    if (16 * mtile + m >= RROWS) continue;
    const int bjh = (sB[m] * Jj + sJ[m]) * Hh;
    const int n   = sN[m];
    float v = acc[i] + bv;
    int t = c >> 9, h = (c >> 6) & 7, d = c & 63;
    if (t == 0) kh [((size_t)(bjh + h) * Nn + n) * Dd + d] = (__bf16)v;
    else        vhr[((size_t)(bjh + h) * Dd + d) * NP + n] = (__bf16)v;
  }
}

// ---------------------------------------------------------------------------
// Kernel 3: dual attention, one wave per (b,j,h, 16-row q-tile).
// S = Q K^T (WMMA) -> LDS fp32; exp -> LDS bf16 (unnormalized);
// O = P V (WMMA on transposed V); 1/sum folded in at accumulation.
// ---------------------------------------------------------------------------
__global__ __launch_bounds__(32) void k_attn(
    const __bf16* __restrict__ qt, const __bf16* __restrict__ kt,
    const __bf16* __restrict__ qh, const __bf16* __restrict__ kh,
    const __bf16* __restrict__ vtr, const __bf16* __restrict__ vhr,
    __bf16* __restrict__ fuse)
{
  __shared__ float  pb[16 * NP];
  __shared__ __bf16 ph[16 * NP];
  const int lane = threadIdx.x & 31;
  const int nl = lane & 15, g = lane >> 4;
  const int bjh = blockIdx.x, tile = blockIdx.y;
  const int b = bjh / (Jj * Hh), jh = bjh % (Jj * Hh);
  const int j = jh / Hh, h = jh % Hh;
  const size_t base  = (size_t)bjh * Nn * Dd;   // q/k layout
  const size_t baseV = (size_t)bjh * Dd * NP;   // transposed v layout
  const float scale = 0.125f;                   // 1/sqrt(64)

  v8f osum[4] = {};

  for (int type = 0; type < 2; ++type) {
    const __bf16* Q  = type ? qh  : qt;
    const __bf16* K  = type ? kh  : kt;
    const __bf16* Vt = type ? vhr : vtr;

    // Q fragments (rows fixed per lane; two 16x32 frags)
    const int nr = min(16 * tile + nl, Nn - 1);
    const __bf16* qrow = Q + base + (size_t)nr * Dd + 8 * g;
    v16bf aq0 = frag16(qrow);
    v16bf aq1 = frag16(qrow + 32);

    // S tiles -> LDS (scaled, OOB cols masked)
    for (int u = 0; u < 16; ++u) {
      const int col = u * 16 + nl;
      const __bf16* krow = K + base + (size_t)min(col, Nn - 1) * Dd + 8 * g;
      v8f sa = {};
      sa = wmma_bf16(aq0, frag16(krow), sa);
      sa = wmma_bf16(aq1, frag16(krow + 32), sa);
#pragma unroll
      for (int i = 0; i < 8; ++i)
        pb[(i + 8 * g) * NP + col] = (col < Nn) ? sa[i] * scale : -1.0e30f;
    }

    // softmax over 256 (row = nl, half g covers 128 cols), vectorized
    const int row = nl, c0 = g * 128;
    float mx = -3.0e38f;
    for (int c = c0; c < c0 + 128; c += 4) {
      const v4f v = *(const v4f*)&pb[row * NP + c];
      mx = fmaxf(mx, fmaxf(fmaxf(v.x, v.y), fmaxf(v.z, v.w)));
    }
    mx = fmaxf(mx, __shfl_xor(mx, 16, 32));
    float sm = 0.f;
    for (int c = c0; c < c0 + 128; c += 4) {
      const v4f v = *(const v4f*)&pb[row * NP + c];
      float e0 = __expf(v.x - mx), e1 = __expf(v.y - mx);
      float e2 = __expf(v.z - mx), e3 = __expf(v.w - mx);
      sm += (e0 + e1) + (e2 + e3);
      v4bf o = { (__bf16)e0, (__bf16)e1, (__bf16)e2, (__bf16)e3 };
      *(v4bf*)&ph[row * NP + c] = o;
    }
    sm += __shfl_xor(sm, 16, 32);
    const float inv = 1.0f / sm;          // lane holds 1/sum for row nl
    float invm[8];
#pragma unroll
    for (int i = 0; i < 8; ++i) invm[i] = __shfl(inv, i + 8 * g, 32);

    // O = P V  (K = 256, 4 output d-tiles); scale rows by invm when adding
    const __bf16* prow = ph + nl * NP + 8 * g;
#pragma unroll
    for (int nt4 = 0; nt4 < 4; ++nt4) {
      const __bf16* vcol = Vt + baseV + (size_t)(nt4 * 16 + nl) * NP + 8 * g;
      v8f oa = {};
#pragma unroll 2
      for (int ks = 0; ks < 8; ++ks) {
        const int off = ks * 32;
        oa = wmma_bf16(frag16(prow + off), frag16(vcol + off), oa);
      }
#pragma unroll
      for (int i = 0; i < 8; ++i) osum[nt4][i] += oa[i] * invm[i];
    }
  }

  // store both-branch sum to fuse (b, n, j, h*64+d) as bf16
#pragma unroll
  for (int nt4 = 0; nt4 < 4; ++nt4)
#pragma unroll
    for (int i = 0; i < 8; ++i) {
      int m  = i + 8 * g;
      int nq = 16 * tile + m;
      if (nq < Nn) {
        int d = nt4 * 16 + nl;
        fuse[((size_t)((b * Nn + nq) * Jj + j)) * Cc + h * 64 + d] =
            (__bf16)osum[nt4][i];
      }
    }
}

// ---------------------------------------------------------------------------
// Kernel 4: out = LayerNorm(x + fuse @ W_proj + b_proj)   rows (b,n,j)
// A-staging uses gfx1250 async global->LDS copy when available.
// ---------------------------------------------------------------------------
__global__ __launch_bounds__(256) void k_proj(
    const __bf16* __restrict__ fuse, const float* __restrict__ x,
    const __bf16* __restrict__ Wt, const float* __restrict__ bp,
    float* __restrict__ out)
{
  __shared__ __bf16 ldsA[16 * 512];
  __shared__ float  ldsO[16 * 512];
  const int tid = threadIdx.x, lane = tid & 31, w = tid >> 5;
  const int nl = lane & 15, g = lane >> 4;
  const int mtile = blockIdx.x;

#if HAVE_ASYNC_LDS
#pragma unroll
  for (int idx = tid; idx < 16 * 64; idx += 256) {
    int r = idx >> 6, c8 = (idx & 63) << 3;
    int gr = min(16 * mtile + r, RROWS - 1);
    async_cp16(fuse + (size_t)gr * Cc + c8, &ldsA[r * 512 + c8]);
  }
#  if __has_builtin(__builtin_amdgcn_s_wait_asynccnt)
  __builtin_amdgcn_s_wait_asynccnt(0);
#  else
  asm volatile("s_wait_asynccnt 0" ::: "memory");
#  endif
#else
  for (int idx = tid; idx < 16 * 64; idx += 256) {
    int r = idx >> 6, c8 = (idx & 63) << 3;
    int gr = min(16 * mtile + r, RROWS - 1);
    *(v8bf*)&ldsA[r * 512 + c8] = *(const v8bf*)&fuse[(size_t)gr * Cc + c8];
  }
#endif
  __syncthreads();

#pragma unroll
  for (int t = 0; t < 4; ++t) {
    const int col0 = (w + 8 * t) * 16;
    const __bf16* wcol = Wt + (size_t)(col0 + nl) * 512;
    const __bf16* arow = ldsA + nl * 512;
    v8f acc = {};
#pragma unroll 4
    for (int ks = 0; ks < 16; ++ks) {
      const int off = ks * 32 + 8 * g;
      acc = wmma_bf16(frag16(arow + off), frag16(wcol + off), acc);
    }
    const float bv = bp[col0 + nl];
#pragma unroll
    for (int i = 0; i < 8; ++i) {
      int m  = i + 8 * g;
      int gr = min(16 * mtile + m, RROWS - 1);
      ldsO[m * 512 + col0 + nl] = acc[i] + bv + x[(size_t)gr * Cc + col0 + nl];
    }
  }
  __syncthreads();

  // layernorm: wave w handles rows 2w, 2w+1; lane covers 16 consecutive cols
  for (int rr = 0; rr < 2; ++rr) {
    const int r = 2 * w + rr;
    float s = 0.f, s2 = 0.f;
#pragma unroll
    for (int c = lane * 16; c < lane * 16 + 16; c += 4) {
      const v4f v = *(const v4f*)&ldsO[r * 512 + c];
      s  += (v.x + v.y) + (v.z + v.w);
      s2 += (v.x*v.x + v.y*v.y) + (v.z*v.z + v.w*v.w);
    }
#pragma unroll
    for (int off = 16; off >= 1; off >>= 1) {
      s  += __shfl_xor(s,  off, 32);
      s2 += __shfl_xor(s2, off, 32);
    }
    const float mean = s * (1.0f / 512.0f);
    const float var  = s2 * (1.0f / 512.0f) - mean * mean;
    const float inv  = rsqrtf(var + 1e-5f);
    const int gr = 16 * mtile + r;
    if (gr < RROWS) {
#pragma unroll
      for (int c = lane * 16; c < lane * 16 + 16; c += 4) {
        const v4f v = *(const v4f*)&ldsO[r * 512 + c];
        v4f o = { (v.x-mean)*inv, (v.y-mean)*inv, (v.z-mean)*inv, (v.w-mean)*inv };
        *(v4f*)&out[(size_t)gr * Cc + c] = o;
      }
    }
  }
}

// ---------------------------------------------------------------------------
extern "C" void kernel_launch(void* const* d_in, const int* in_sizes, int n_in,
                              void* d_out, int out_size, void* d_ws, size_t ws_size,
                              hipStream_t stream) {
  (void)in_sizes; (void)n_in; (void)out_size; (void)ws_size;
  const float* x    = (const float*)d_in[0];
  const float* Wqkv = (const float*)d_in[1];
  const float* bqkv = (const float*)d_in[2];
  const float* Wq   = (const float*)d_in[3];
  const float* bq   = (const float*)d_in[4];
  const float* Wkv  = (const float*)d_in[5];
  const float* bkv  = (const float*)d_in[6];
  const float* Wp   = (const float*)d_in[7];
  const float* bp   = (const float*)d_in[8];
  float* out = (float*)d_out;

  const size_t per  = (size_t)Bb * Jj * Hh * Nn * Dd;   // 16,920,576
  const size_t vper = (size_t)Bb * Jj * Hh * Dd * NP;   // 17,825,792
  __bf16* qt   = (__bf16*)d_ws;
  __bf16* kt   = qt + per;
  __bf16* qh   = kt + per;
  __bf16* kh   = qh + per;
  __bf16* fuse = kh + per;
  __bf16* vtr  = fuse + per;
  __bf16* vhr  = vtr + vper;
  __bf16* Wt   = vhr + vper;                            // 3584 cols x 512 k

  // weight transpose+convert (Wt col ranges: [0,1536) qkv, [1536,2048) q,
  // [2048,3072) kv, [3072,3584) proj)
  k_wt<<<(1536*512 + 255)/256, 256, 0, stream>>>(Wqkv, Wt, 1536);
  k_wt<<<( 512*512 + 255)/256, 256, 0, stream>>>(Wq,   Wt + (size_t)1536*512, 512);
  k_wt<<<(1024*512 + 255)/256, 256, 0, stream>>>(Wkv,  Wt + (size_t)2048*512, 1024);
  k_wt<<<( 512*512 + 255)/256, 256, 0, stream>>>(Wp,   Wt + (size_t)3072*512, 512);

  k_qkv<<<dim3(MT, 16, 1), 256, 0, stream>>>(x, Wt, bqkv, bq, qt, kt, vtr, qh);
  k_kv<<<dim3(MT, 8, 1), 256, 0, stream>>>(x, Wt + (size_t)2048*512, bkv, kh, vhr);
  k_attn<<<dim3(Bb * Jj * Hh, 16, 1), 32, 0, stream>>>(qt, kt, qh, kh, vtr, vhr, fuse);
  k_proj<<<dim3(MT, 1, 1), 256, 0, stream>>>(fuse, x, Wt + (size_t)3072*512, bp, out);
}
